// Encoder_47622597378544
// MI455X (gfx1250) — matrix operand order repaired and checked by reading, not compiled, not run on hardware
//
#include <hip/hip_runtime.h>
#include <cstdint>
#include <cstddef>

// ---------------------------------------------------------------------------
// CDNA5 (gfx1250) bidirectional GRU encoder.
//  Kernel 1: fused embedding-gather + input projection GEMM (WMMA f16->f32)
//  Kernel 2: sequential GRU scan, one workgroup per direction, U cached in
//            VGPRs as WMMA B-operands, h/rec staged in LDS.
// ---------------------------------------------------------------------------

typedef __attribute__((ext_vector_type(16))) _Float16 v16h;
typedef __attribute__((ext_vector_type(8)))  _Float16 v8h;
typedef __attribute__((ext_vector_type(8)))  float    v8f;

union V16H { v16h v; v8h h2[2]; _Float16 e[16]; };

#define WMMA_F32_F16(a, b, c) \
  __builtin_amdgcn_wmma_f32_16x16x32_f16(false, (a), false, (b), (short)0, (c), false, false)

constexpr int Tn = 512;   // sequence length
constexpr int Bn = 64;    // batch
constexpr int En = 256;   // embedding dim
constexpr int Un = 256;   // hidden dim
constexpr int G3 = 768;   // 3*U

// ---------------------------------------------------------------------------
// Kernel 1: xp[d][t][b][:] = emb[x[b][t]] @ W_d + bias_d   (layout [T][B][3U])
// 128x128 output tile per 256-thread workgroup, K=256 staged in LDS as f16.
// ---------------------------------------------------------------------------
__global__ __launch_bounds__(256) void xproj_kernel(
    const int* __restrict__ x, const float* __restrict__ emb,
    const float* __restrict__ Wf, const float* __restrict__ bf,
    const float* __restrict__ Wb, const float* __restrict__ bb,
    float* __restrict__ xpF, float* __restrict__ xpB)
{
  extern __shared__ char smem[];
  constexpr int AP = 264;  // A tile pitch (f16 elems), 16B-aligned rows
  constexpr int WP = 264;  // W^T tile pitch
  _Float16* Alds = (_Float16*)smem;                      // [128][AP]  rows x K
  _Float16* WT   = (_Float16*)(smem + 128 * AP * 2);     // [128][WP]  cols x K

  const int dir = blockIdx.z;
  const float* W    = dir ? Wb : Wf;
  const float* bias = dir ? bb : bf;            // row 0 = input bias
  float* xp         = dir ? xpB : xpF;

  const int rowBase = blockIdx.y * 128;         // over T*B = 32768 rows
  const int colBase = blockIdx.x * 128;         // over 3U = 768 cols
  const int tid = threadIdx.x;

  // Stage A: gather 128 embedding rows -> LDS (f16). 2 threads per row.
  {
    const int rl = tid >> 1, half = tid & 1;
    const int rg = rowBase + rl;
    const int t = rg >> 6, b = rg & 63;
    const int tok = x[b * Tn + t];
    const float4* src = (const float4*)(emb + (size_t)tok * En + half * 128);
    _Float16* dst = Alds + rl * AP + half * 128;
    #pragma unroll
    for (int i = 0; i < 32; ++i) {
      float4 f = src[i];
      dst[i * 4 + 0] = (_Float16)f.x;
      dst[i * 4 + 1] = (_Float16)f.y;
      dst[i * 4 + 2] = (_Float16)f.z;
      dst[i * 4 + 3] = (_Float16)f.w;
    }
  }
  // Stage B: W tile (256 x 128) transposed into LDS as [col][k] f16.
  for (int idx = tid; idx < 256 * 128; idx += 256) {
    const int k = idx >> 7, c = idx & 127;
    WT[c * WP + k] = (_Float16)W[(size_t)k * G3 + colBase + c];
  }
  __syncthreads();

  const int lane  = tid & 31;
  const int wave  = tid >> 5;
  const int mlane = lane & 15;
  const int kbA = (lane < 16) ? 0 : 8;    // A operand: two 8-elem K chunks
  const int kbB = (lane < 16) ? 0 : 8;    // B operand: two 8-elem K chunks
  const int rt0 = (wave & 3) * 2;         // 2 row tiles
  const int ct0 = (wave >> 2) * 4;        // 4 col tiles

  const v8f zacc = {};
  v8f acc[2][4];
  #pragma unroll
  for (int i = 0; i < 2; ++i)
    #pragma unroll
    for (int j = 0; j < 4; ++j) acc[i][j] = zacc;

  #pragma unroll
  for (int kc = 0; kc < 8; ++kc) {
    V16H a[2], bm[4];
    #pragma unroll
    for (int i = 0; i < 2; ++i) {
      const _Float16* ap = Alds + ((rt0 + i) * 16 + mlane) * AP + kc * 32 + kbA;
      a[i].h2[0] = *(const v8h*)ap;
      a[i].h2[1] = *(const v8h*)(ap + 16);
    }
    #pragma unroll
    for (int j = 0; j < 4; ++j) {
      const _Float16* bp = WT + ((ct0 + j) * 16 + mlane) * WP + kc * 32 + kbB;
      bm[j].h2[0] = *(const v8h*)bp;
      bm[j].h2[1] = *(const v8h*)(bp + 16);
    }
    #pragma unroll
    for (int i = 0; i < 2; ++i)
      #pragma unroll
      for (int j = 0; j < 4; ++j)
        acc[i][j] = WMMA_F32_F16(a[i].v, bm[j].v, acc[i][j]);
  }

  // Epilogue: add bias, store f32 xp.
  const int mofs = (lane >> 4) << 3;
  #pragma unroll
  for (int i = 0; i < 2; ++i) {
    #pragma unroll
    for (int j = 0; j < 4; ++j) {
      const int colG = colBase + (ct0 + j) * 16 + mlane;
      const float bv = bias[colG];
      #pragma unroll
      for (int v = 0; v < 8; ++v) {
        const int rowG = rowBase + (rt0 + i) * 16 + v + mofs;
        xp[(size_t)rowG * G3 + colG] = acc[i][j][v] + bv;
      }
    }
  }
}

// ---------------------------------------------------------------------------
// Kernel 2: GRU scan. grid = {fwd, bwd}; 768 threads = 24 waves.
// Each wave owns a 32-col slice of U (f16 B-operands resident in VGPRs).
// ---------------------------------------------------------------------------
__global__ __launch_bounds__(768) void gru_scan_kernel(
    const int* __restrict__ x, const float* __restrict__ Uf,
    const float* __restrict__ bf, const float* __restrict__ Ub,
    const float* __restrict__ bb, const float* __restrict__ xpF,
    const float* __restrict__ xpB, float* __restrict__ out)
{
  extern __shared__ char smem[];
  constexpr int RECP = 776;   // rec pitch (f32)
  constexpr int HP16 = 272;   // h f16 pitch
  constexpr int H32P = 264;   // h f32 pitch
  float*    rec = (float*)smem;                                      // [64][RECP]
  _Float16* h16 = (_Float16*)(smem + 64 * RECP * 4);                 // [64][HP16]
  float*    h32 = (float*)(smem + 64 * RECP * 4 + 64 * HP16 * 2);    // [64][H32P]

  const int dir = blockIdx.x;
  const float* Umat = dir ? Ub : Uf;
  const float* brec = (dir ? bb : bf) + G3;   // row 1 = recurrent bias
  const float* xp   = dir ? xpB : xpF;

  const int tid   = threadIdx.x;
  const int lane  = tid & 31;
  const int wave  = tid >> 5;           // 0..23
  const int nlane = lane & 15;
  const int kbA = (lane < 16) ? 0 : 16;
  const int kbB = (lane < 16) ? 0 : 16;
  const int colBase = wave * 32;        // this wave's columns of 768

  // Preload this wave's U slice into registers (16 B-operands = 128 VGPRs).
  V16H Breg[8][2];
  #pragma unroll
  for (int kc = 0; kc < 8; ++kc) {
    #pragma unroll
    for (int ctl = 0; ctl < 2; ++ctl) {
      const float* up = Umat + (size_t)(kc * 32 + kbB) * G3 + colBase + ctl * 16 + nlane;
      #pragma unroll
      for (int i = 0; i < 16; ++i)
        Breg[kc][ctl].e[i] = (_Float16)up[(size_t)i * G3];
    }
  }

  // h = 0
  for (int idx = tid; idx < Bn * Un; idx += 768) {
    const int b = idx >> 8, j = idx & 255;
    h32[b * H32P + j] = 0.f;
    h16[b * HP16 + j] = (_Float16)0.f;
  }
  __syncthreads();

  const v8f zacc = {};
  for (int s = 0; s < Tn; ++s) {
    const int t = dir ? (Tn - 1 - s) : s;

    // Phase A: rec = h @ U via WMMA; 4 row tiles x 2 col tiles per wave.
    v8f acc[4][2];
    #pragma unroll
    for (int rt = 0; rt < 4; ++rt) { acc[rt][0] = zacc; acc[rt][1] = zacc; }
    #pragma unroll
    for (int kc = 0; kc < 8; ++kc) {
      #pragma unroll
      for (int rt = 0; rt < 4; ++rt) {
        V16H a;
        const _Float16* ap = h16 + (rt * 16 + nlane) * HP16 + kc * 32 + kbA;
        a.h2[0] = *(const v8h*)ap;
        a.h2[1] = *(const v8h*)(ap + 8);
        acc[rt][0] = WMMA_F32_F16(a.v, Breg[kc][0].v, acc[rt][0]);
        acc[rt][1] = WMMA_F32_F16(a.v, Breg[kc][1].v, acc[rt][1]);
      }
    }
    const int mofs = (lane >> 4) << 3;
    #pragma unroll
    for (int rt = 0; rt < 4; ++rt)
      #pragma unroll
      for (int ctl = 0; ctl < 2; ++ctl)
        #pragma unroll
        for (int v = 0; v < 8; ++v)
          rec[(rt * 16 + v + mofs) * RECP + colBase + ctl * 16 + nlane] = acc[rt][ctl][v];
    __syncthreads();

    // Prefetch next timestep's projections (global_prefetch_b8).
    {
      const int tn = dir ? (t - 1) : (t + 1);
      if (tn >= 0 && tn < Tn)
        __builtin_prefetch(xp + (size_t)(tn * Bn) * G3 + (tid << 6), 0, 0);
    }

    // Phase B: gates + state update (elementwise over 64x256).
    const float* xpt = xp + (size_t)(t * Bn) * G3;
    for (int idx = tid; idx < Bn * Un; idx += 768) {
      const int b = idx >> 8, j = idx & 255;
      const float xz = xpt[b * G3 + j];
      const float xr = xpt[b * G3 + 256 + j];
      const float xh = xpt[b * G3 + 512 + j];
      const float rz = rec[b * RECP + j]       + brec[j];
      const float rr = rec[b * RECP + 256 + j] + brec[256 + j];
      const float rh = rec[b * RECP + 512 + j] + brec[512 + j];
      const float h  = h32[b * H32P + j];
      const float z  = 1.f / (1.f + __expf(-(xz + rz)));
      const float r  = 1.f / (1.f + __expf(-(xr + rr)));
      const float hh = tanhf(xh + r * rh);
      float hn = z * h + (1.f - z) * hh;
      if (x[b * Tn + t] == 0) hn = h;           // masked step: carry state
      h32[b * H32P + j] = hn;
      h16[b * HP16 + j] = (_Float16)hn;
      out[((size_t)b * Tn + t) * (2 * Un) + dir * Un + j] = hn;
    }
    __syncthreads();
  }

  // Final hidden states h_f / h_b appended after out.
  float* hT = out + (size_t)Bn * Tn * 2 * Un + (size_t)dir * Bn * Un;
  for (int idx = tid; idx < Bn * Un; idx += 768) {
    const int b = idx >> 8, j = idx & 255;
    hT[b * Un + j] = h32[b * H32P + j];
  }
}

// ---------------------------------------------------------------------------
extern "C" void kernel_launch(void* const* d_in, const int* in_sizes, int n_in,
                              void* d_out, int out_size, void* d_ws, size_t ws_size,
                              hipStream_t stream) {
  (void)in_sizes; (void)n_in; (void)out_size; (void)ws_size;
  const int*   x   = (const int*)d_in[0];
  const float* emb = (const float*)d_in[1];
  const float* Wf  = (const float*)d_in[2];
  const float* Uf  = (const float*)d_in[3];
  const float* bf  = (const float*)d_in[4];
  const float* Wb  = (const float*)d_in[5];
  const float* Ub  = (const float*)d_in[6];
  const float* bb  = (const float*)d_in[7];
  float* out = (float*)d_out;

  const size_t xpElems = (size_t)Tn * Bn * G3;     // 25,165,824 f32 each dir
  float* xpF = (float*)d_ws;
  float* xpB = xpF + xpElems;

  const size_t lds1 = (size_t)128 * 264 * 2 * 2;                               // ~135 KB
  const size_t lds2 = (size_t)64 * 776 * 4 + 64 * 272 * 2 + 64 * 264 * 4;      // ~294 KB

  (void)hipFuncSetAttribute((const void*)xproj_kernel,
                            hipFuncAttributeMaxDynamicSharedMemorySize, (int)lds1);
  (void)hipFuncSetAttribute((const void*)gru_scan_kernel,
                            hipFuncAttributeMaxDynamicSharedMemorySize, (int)lds2);

  dim3 g1(G3 / 128, (Tn * Bn) / 128, 2);   // 6 x 256 x 2 workgroups
  xproj_kernel<<<g1, 256, lds1, stream>>>(x, emb, Wf, bf, Wb, bb, xpF, xpB);
  gru_scan_kernel<<<dim3(2), 768, lds2, stream>>>(x, Uf, bf, Ub, bb, xpF, xpB, out);
}
